// ImportancePropagationLayer_3478923510357
// MI455X (gfx1250) — matrix operand order repaired and checked by reading, not compiled, not run on hardware
//
#include <hip/hip_runtime.h>

typedef __attribute__((ext_vector_type(16))) _Float16 v16h;
typedef __attribute__((ext_vector_type(8)))  float    v8f;

__device__ __forceinline__ v8f wmma16(v16h a, v16h b, v8f c) {
  // D = A(16x32 f16) * B(32x16 f16) + C(16x16 f32)
  return __builtin_amdgcn_wmma_f32_16x16x32_f16(false, a, false, b, (short)0, c,
                                                false, false);
}

// Stage CHUNKS*4KB global->LDS with CDNA5 async copies (256-thread blocks).
// Low 32 bits of a generic LDS pointer are the LDS byte offset (aperture rule).
template <int CHUNKS>
__device__ __forceinline__ void stage_async(const void* g, void* lds) {
  unsigned lds0 = (unsigned)(size_t)lds;
  const char* gp = (const char*)g;
#pragma unroll
  for (int j = 0; j < CHUNKS; ++j) {
    unsigned boff = (threadIdx.x + 256u * (unsigned)j) * 16u;
    asm volatile("global_load_async_to_lds_b128 %0, %1, off"
                 :: "v"(lds0 + boff), "v"(gp + boff) : "memory");
  }
  asm volatile("s_wait_asynccnt 0x0" ::: "memory");
  __syncthreads();
}

// ---------------------------------------------------------------------------
// Pack f32 weight W[k, n] (row-major, ldW cols, 8 N-tiles of 16) into WMMA
// B-fragments (f16): frag[((tile*kchunks+kc)*32+lane)*16+i],
// k = kc*32 + (lane/16)*16 + i, n = tile*16 + lane%16.
// ---------------------------------------------------------------------------
__global__ void pack_b_frag(const float* __restrict__ W, _Float16* __restrict__ frag,
                            int kchunks, int ldW) {
  int t = blockIdx.x * blockDim.x + threadIdx.x;
  int total = 8 * kchunks * 32 * 16;
  if (t >= total) return;
  int i    = t & 15;
  int lane = (t >> 4) & 31;
  int rest = t >> 9;
  int kc   = rest % kchunks;
  int tile = rest / kchunks;
  int k = kc * 32 + (lane >> 4) * 16 + i;
  int n = tile * 16 + (lane & 15);
  frag[t] = (_Float16)W[k * ldW + n];
}

// ---------------------------------------------------------------------------
// Edge stage: msg = relu(x[src] + edge_attr @ We + be); agg[dst] += msg.
// Persistent waves; We fragments in LDS (async-staged once per block).
// ---------------------------------------------------------------------------
__global__ void __launch_bounds__(256, 2)
edge_msg_wmma(const float* __restrict__ edge_attr,      // [E,32]
              const long long* __restrict__ edge_index, // [2,E]
              const float* __restrict__ x,              // [N,128]
              const _Float16* __restrict__ We_frag,     // [8][32][16]
              const float* __restrict__ be,             // [128]
              float* __restrict__ agg,                  // [N,128]
              int E, int ntiles) {
  __shared__ __attribute__((aligned(32))) _Float16 ldsWe[8 * 32 * 16];  // 8 KB
  stage_async<2>(We_frag, ldsWe);

  int lane = threadIdx.x & 31;
  int hf   = lane >> 4;
  int m    = lane & 15;

  float biasv[8];
#pragma unroll
  for (int nt = 0; nt < 8; ++nt) biasv[nt] = be[nt * 16 + m];

  const char* xb = (const char*)(x + m);    // per-lane column base
  char*       ab = (char*)(agg + m);
  const int*  ei32 = (const int*)edge_index;  // low word of little-endian int64

  int gw = blockIdx.x * (blockDim.x >> 5) + (threadIdx.x >> 5);
  int nw = gridDim.x * (blockDim.x >> 5);

  for (int tile = gw; tile < ntiles; tile += nw) {
    int base = tile * 16;

    // A fragment: 16x32 f32 tile of edge_attr -> f16, contiguous k = 16*hf + s order (as pack_b_frag).
    const float* arow = edge_attr + (size_t)(base + m) * 32;
    float4 q0 = *(const float4*)(arow + hf * 16 + 0);
    float4 q1 = *(const float4*)(arow + hf * 16 + 4);
    float4 q2 = *(const float4*)(arow + hf * 16 + 8);
    float4 q3 = *(const float4*)(arow + hf * 16 + 12);
    v16h a;
    a[0]=(_Float16)q0.x; a[1]=(_Float16)q0.y; a[2]=(_Float16)q0.z; a[3]=(_Float16)q0.w;
    a[4]=(_Float16)q1.x; a[5]=(_Float16)q1.y; a[6]=(_Float16)q1.z; a[7]=(_Float16)q1.w;
    a[8]=(_Float16)q2.x; a[9]=(_Float16)q2.y; a[10]=(_Float16)q2.z; a[11]=(_Float16)q2.w;
    a[12]=(_Float16)q3.x; a[13]=(_Float16)q3.y; a[14]=(_Float16)q3.z; a[15]=(_Float16)q3.w;

    // src/dst ids for the 8 rows this lane's C elements map to (M = r + 8*hf).
    // Half-wave lanes read identical addresses -> single coalesced L0 hit.
    unsigned sOff[8], dOff[8];                 // unsigned byte offsets (row*512)
#pragma unroll
    for (int r = 0; r < 8; ++r) {
      int e2 = (base + 8 * hf + r) * 2;
      sOff[r] = (unsigned)ei32[e2] << 9;
      dOff[r] = (unsigned)ei32[(size_t)E * 2 + e2] << 9;
    }

#pragma unroll
    for (int nt = 0; nt < 8; ++nt) {
      v16h b = *(const v16h*)(ldsWe + ((nt * 32 + lane) << 4));
      float xv[8];
#pragma unroll
      for (int r = 0; r < 8; ++r)              // gathers issued before WMMA
        xv[r] = *(const float*)(xb + sOff[r] + nt * 64);
      v8f c;
#pragma unroll
      for (int r = 0; r < 8; ++r) c[r] = biasv[nt];
      c = wmma16(a, b, c);
#pragma unroll
      for (int r = 0; r < 8; ++r) {
        float msg = c[r] + xv[r];
        msg = msg > 0.f ? msg : 0.f;
        atomicAdd((float*)(ab + dOff[r] + nt * 64), msg);
      }
    }
  }
}

// Scalar tail for E % 16 != 0 (not taken for E = 800000).
__global__ void edge_scalar_tail(const float* __restrict__ ea,
                                 const long long* __restrict__ ei,
                                 const float* __restrict__ x,
                                 const float* __restrict__ We,
                                 const float* __restrict__ be,
                                 float* __restrict__ agg, int E, int start) {
  int t = blockIdx.x * blockDim.x + threadIdx.x;
  int nE = E - start;
  if (t >= nE * 128) return;
  int e = start + t / 128;
  int ch = t & 127;
  float s = be[ch];
  for (int k = 0; k < 32; ++k) s += ea[(size_t)e * 32 + k] * We[k * 128 + ch];
  int sn = (int)ei[e], dn = (int)ei[E + e];
  float msg = x[(size_t)sn * 128 + ch] + s;
  msg = msg > 0.f ? msg : 0.f;
  atomicAdd(agg + (size_t)dn * 128 + ch, msg);
}

// ---------------------------------------------------------------------------
// conv_out = (x + agg) @ Wc + bc. conv_out aliases agg (xa): a tile's rows
// are read fully before being written, and rows are tile-exclusive.
// ---------------------------------------------------------------------------
__global__ void __launch_bounds__(256, 2)
conv_gemm(const float* __restrict__ x,
          float* xa,                             // in agg / out conv
          const _Float16* __restrict__ Wc_frag,  // [8][4][32][16]
          const float* __restrict__ bc,
          int ntiles) {
  __shared__ __attribute__((aligned(32))) _Float16 ldsW[8 * 4 * 32 * 16];  // 32 KB
  stage_async<8>(Wc_frag, ldsW);

  int lane = threadIdx.x & 31;
  int hf = lane >> 4, m = lane & 15;
  const float* bcm = bc + m;

  int gw = blockIdx.x * (blockDim.x >> 5) + (threadIdx.x >> 5);
  int nw = gridDim.x * (blockDim.x >> 5);

  for (int tile = gw; tile < ntiles; tile += nw) {
    int base = tile * 16;
    const float* xrow = x  + (size_t)(base + m) * 128;
    const float* grow = xa + (size_t)(base + m) * 128;

    v16h a[4];
#pragma unroll
    for (int kc = 0; kc < 4; ++kc) {
      int k0 = kc * 32 + hf * 16;
      float4 q0 = *(const float4*)(xrow + k0);
      float4 q1 = *(const float4*)(xrow + k0 + 4);
      float4 q2 = *(const float4*)(xrow + k0 + 8);
      float4 q3 = *(const float4*)(xrow + k0 + 12);
      float4 g0 = *(const float4*)(grow + k0);
      float4 g1 = *(const float4*)(grow + k0 + 4);
      float4 g2 = *(const float4*)(grow + k0 + 8);
      float4 g3 = *(const float4*)(grow + k0 + 12);
      a[kc][0]=(_Float16)(q0.x+g0.x); a[kc][1]=(_Float16)(q0.y+g0.y);
      a[kc][2]=(_Float16)(q0.z+g0.z); a[kc][3]=(_Float16)(q0.w+g0.w);
      a[kc][4]=(_Float16)(q1.x+g1.x); a[kc][5]=(_Float16)(q1.y+g1.y);
      a[kc][6]=(_Float16)(q1.z+g1.z); a[kc][7]=(_Float16)(q1.w+g1.w);
      a[kc][8]=(_Float16)(q2.x+g2.x); a[kc][9]=(_Float16)(q2.y+g2.y);
      a[kc][10]=(_Float16)(q2.z+g2.z); a[kc][11]=(_Float16)(q2.w+g2.w);
      a[kc][12]=(_Float16)(q3.x+g3.x); a[kc][13]=(_Float16)(q3.y+g3.y);
      a[kc][14]=(_Float16)(q3.z+g3.z); a[kc][15]=(_Float16)(q3.w+g3.w);
    }

    // (base + r + 8*hf)*128 + m  =  B0 + r*128 + nt*16  (immediates)
    float* pob = xa + (size_t)(base + 8 * hf) * 128 + m;

#pragma unroll
    for (int nt = 0; nt < 8; ++nt) {
      float bias = bcm[nt * 16];
      v8f c;
#pragma unroll
      for (int r = 0; r < 8; ++r) c[r] = bias;
#pragma unroll
      for (int kc = 0; kc < 4; ++kc) {
        v16h b = *(const v16h*)(ldsW + (((nt * 4 + kc) * 32 + lane) << 4));
        c = wmma16(a[kc], b, c);
      }
#pragma unroll
      for (int r = 0; r < 8; ++r) pob[r * 128 + nt * 16] = c[r];
    }
  }
}

// ---------------------------------------------------------------------------
// gate = sigmoid(conv @ Wg[:128] + imp * Wg[128] + bg);
// out  = gate*conv + (1-gate)*x;  prop = out @ Wp + bp (fused, in-register).
// ---------------------------------------------------------------------------
__global__ void __launch_bounds__(256, 2)
gate_out(const float* __restrict__ x,
         const float* __restrict__ conv,
         const float* __restrict__ importance,
         const _Float16* __restrict__ Wg_frag, // [8][4][32][16]
         const float* __restrict__ WgLast,     // Wg row 128
         const float* __restrict__ bg,
         const float* __restrict__ Wp,         // [128]
         const float* __restrict__ bp,         // [1]
         float* __restrict__ out,
         float* __restrict__ prop, int ntiles) {
  __shared__ __attribute__((aligned(32))) _Float16 ldsW[8 * 4 * 32 * 16];  // 32 KB
  stage_async<8>(Wg_frag, ldsW);

  int lane = threadIdx.x & 31;
  int hf = lane >> 4, m = lane & 15;
  const float* bgm = bg + m;
  const float* wlm = WgLast + m;
  const float* wpm = Wp + m;
  float bp0 = bp[0];

  int gw = blockIdx.x * (blockDim.x >> 5) + (threadIdx.x >> 5);
  int nw = gridDim.x * (blockDim.x >> 5);

  for (int tile = gw; tile < ntiles; tile += nw) {
    int base = tile * 16;
    const float* crow = conv + (size_t)(base + m) * 128;

    v16h a[4];
#pragma unroll
    for (int kc = 0; kc < 4; ++kc) {
      int k0 = kc * 32 + hf * 16;
      float4 q0 = *(const float4*)(crow + k0);
      float4 q1 = *(const float4*)(crow + k0 + 4);
      float4 q2 = *(const float4*)(crow + k0 + 8);
      float4 q3 = *(const float4*)(crow + k0 + 12);
      a[kc][0]=(_Float16)q0.x; a[kc][1]=(_Float16)q0.y; a[kc][2]=(_Float16)q0.z; a[kc][3]=(_Float16)q0.w;
      a[kc][4]=(_Float16)q1.x; a[kc][5]=(_Float16)q1.y; a[kc][6]=(_Float16)q1.z; a[kc][7]=(_Float16)q1.w;
      a[kc][8]=(_Float16)q2.x; a[kc][9]=(_Float16)q2.y; a[kc][10]=(_Float16)q2.z; a[kc][11]=(_Float16)q2.w;
      a[kc][12]=(_Float16)q3.x; a[kc][13]=(_Float16)q3.y; a[kc][14]=(_Float16)q3.z; a[kc][15]=(_Float16)q3.w;
    }

    // (base + r + 8*hf)*128 + m = B0 + r*128 + nt*16 (immediates)
    size_t B0 = (size_t)(base + 8 * hf) * 128 + m;
    const float* pcb = conv + B0;
    const float* pxb = x    + B0;
    float*       pob = out  + B0;
    const float* impb = importance + base + 8 * hf;

    float sloc[8];
#pragma unroll
    for (int r = 0; r < 8; ++r) sloc[r] = 0.f;

#pragma unroll
    for (int nt = 0; nt < 8; ++nt) {
      float bias = bgm[nt * 16];
      float wlv  = wlm[nt * 16];
      float wpv  = wpm[nt * 16];
      v8f c;
#pragma unroll
      for (int r = 0; r < 8; ++r) c[r] = bias;
#pragma unroll
      for (int kc = 0; kc < 4; ++kc) {
        v16h b = *(const v16h*)(ldsW + (((nt * 4 + kc) * 32 + lane) << 4));
        c = wmma16(a[kc], b, c);
      }
#pragma unroll
      for (int r = 0; r < 8; ++r) {
        float gp = c[r] + impb[r] * wlv;
        float gate = 1.f / (1.f + __expf(-gp));
        float co = pcb[r * 128 + nt * 16];
        float xv = pxb[r * 128 + nt * 16];
        float o = gate * co + (1.f - gate) * xv;
        pob[r * 128 + nt * 16] = o;
        sloc[r] += o * wpv;
      }
    }

    // prop: reduce each row's partial dot across the 16 lanes of its half.
#pragma unroll
    for (int r = 0; r < 8; ++r) {
      float s = sloc[r];
      s += __shfl_xor(s, 1); s += __shfl_xor(s, 2);
      s += __shfl_xor(s, 4); s += __shfl_xor(s, 8);
      if (m == 0) prop[base + r + 8 * hf] = s + bp0;
    }
  }
}

// ---------------------------------------------------------------------------
extern "C" void kernel_launch(void* const* d_in, const int* in_sizes, int n_in,
                              void* d_out, int out_size, void* d_ws, size_t ws_size,
                              hipStream_t stream) {
  const float*     x   = (const float*)d_in[0];
  const long long* ei  = (const long long*)d_in[1];
  const float*     ea  = (const float*)d_in[2];
  const float*     imp = (const float*)d_in[3];
  const float*     We  = (const float*)d_in[4];
  const float*     be  = (const float*)d_in[5];
  const float*     Wc  = (const float*)d_in[6];
  const float*     bc  = (const float*)d_in[7];
  const float*     Wg  = (const float*)d_in[8];
  const float*     bg  = (const float*)d_in[9];
  const float*     Wp  = (const float*)d_in[10];
  const float*     bp  = (const float*)d_in[11];

  const int N = in_sizes[0] / 128;   // 50000
  const int E = in_sizes[2] / 32;    // 800000

  float* outF  = (float*)d_out;
  float* propF = outF + (size_t)N * 128;

  char* ws = (char*)d_ws;
  float*    agg = (float*)ws;                              // N*128 f32 (aliased conv_out)
  _Float16* WeF = (_Float16*)(ws + (size_t)N * 128 * 4);   // 8*32*16
  _Float16* WcF = WeF + 8 * 32 * 16;                       // 8*4*32*16
  _Float16* WgF = WcF + 8 * 4 * 32 * 16;                   // 8*4*32*16

  hipMemsetAsync(agg, 0, (size_t)N * 128 * sizeof(float), stream);

  pack_b_frag<<<(8 * 1 * 32 * 16 + 255) / 256, 256, 0, stream>>>(We, WeF, 1, 128);
  pack_b_frag<<<(8 * 4 * 32 * 16 + 255) / 256, 256, 0, stream>>>(Wc, WcF, 4, 128);
  pack_b_frag<<<(8 * 4 * 32 * 16 + 255) / 256, 256, 0, stream>>>(Wg, WgF, 4, 128);

  int etiles = E / 16;
  edge_msg_wmma<<<1024, 256, 0, stream>>>(ea, ei, x, WeF, be, agg, E, etiles);
  int tail = E - etiles * 16;
  if (tail > 0)
    edge_scalar_tail<<<(tail * 128 + 255) / 256, 256, 0, stream>>>(ea, ei, x, We, be, agg,
                                                                   E, etiles * 16);

  int ntiles = N / 16;  // N = 50000 is a multiple of 16
  conv_gemm<<<256, 256, 0, stream>>>(x, agg, WcF, bc, ntiles);
  gate_out<<<256, 256, 0, stream>>>(x, agg, imp, WgF, Wg + 128 * 128, bg, Wp, bp,
                                    outF, propF, ntiles);
}
